// GCL_67851893342731
// MI455X (gfx1250) — matrix operand order repaired and checked by reading, not compiled, not run on hardware
//
#include <hip/hip_runtime.h>
#include <hip/hip_bf16.h>

// ---------------------------------------------------------------------------
// EGNN GCL layer for MI455X (gfx1250, wave32, WMMA).
// All GEMMs use v_wmma_f32_16x16x32_f16 (f16 A/B, f32 accum).
// ---------------------------------------------------------------------------

typedef __attribute__((ext_vector_type(16))) _Float16 v16h;
typedef __attribute__((ext_vector_type(8)))  _Float16 v8h;
typedef __attribute__((ext_vector_type(8)))  float    v8f;

#define DF    128          // node feature dim (in == out == hidden)
#define DE    16           // edge attr dim
#define KP1   288          // 2*DF + DE = 272, padded to multiple of 32
#define INV_NORM (1.0f / 100.0f)
#define BN_EPS 1e-5f

__device__ __forceinline__ float scaled_silu(float x) {
    // silu(x) / 0.6
    return (x / (1.0f + __expf(-x))) * (1.0f / 0.6f);
}

// A fragment, 16x32 f16 (ISA 7.12.2): lane group g = lane>>4 ;
// VGPR v holds K = (v&4 ? 16:0) + g*8 + 2*(v&3)  (pairs).
// => per lane: two contiguous 8-half (16B) loads at k0+g*8 and k0+g*8+16.
__device__ __forceinline__ v16h load_a_frag(const _Float16* __restrict__ base,
                                            int row, int ldk, int k0, int lane) {
    int g = (lane >> 4) & 1;
    const _Float16* p = base + row * ldk + k0 + g * 8;
    v8h lo = *(const v8h*)(p);
    v8h hi = *(const v8h*)(p + 16);
    v16h f;
#pragma unroll
    for (int i = 0; i < 8; ++i) { f[i] = lo[i]; f[i + 8] = hi[i]; }
    return f;
}

// B fragment, 32x16 f16: same K order as A fragment (k0+g*8 and k0+g*8+16).
// Weights stored [out][K] row-major so
// this is two contiguous 8-half (16B) loads per lane.
__device__ __forceinline__ v16h load_b_frag(const _Float16* __restrict__ W,
                                            int nrow, int ldk, int k0, int lane) {
    int g = (lane >> 4) & 1;
    const _Float16* p = W + nrow * ldk + k0 + g * 8;
    v8h lo = *(const v8h*)(p);
    v8h hi = *(const v8h*)(p + 16);
    v16h f;
#pragma unroll
    for (int i = 0; i < 8; ++i) { f[i] = lo[i]; f[i + 8] = hi[i]; }
    return f;
}

// ---------------------------------------------------------------------------
// Kernel 0: convert weights f32 -> f16 (pad ew1 K to 288 with zeros)
// ---------------------------------------------------------------------------
__global__ __launch_bounds__(256) void cvt_weights_kernel(
    const float* __restrict__ ew1, const float* __restrict__ ew2,
    const float* __restrict__ nw1, const float* __restrict__ nw2,
    _Float16* __restrict__ ew1h, _Float16* __restrict__ ew2h,
    _Float16* __restrict__ nw1h, _Float16* __restrict__ nw2h) {
    int idx = blockIdx.x * 256 + threadIdx.x;
    if (idx < DF * KP1) {
        int n = idx / KP1, k = idx - n * KP1;
        ew1h[idx] = (_Float16)((k < 2 * DF + DE) ? ew1[n * (2 * DF + DE) + k] : 0.0f);
    }
    if (idx < DF * DF) {
        ew2h[idx] = (_Float16)ew2[idx];
        nw2h[idx] = (_Float16)nw2[idx];
    }
    if (idx < DF * 2 * DF) nw1h[idx] = (_Float16)nw1[idx];
}

// ---------------------------------------------------------------------------
// Kernel 1: zero BN accumulators + agg (contiguous region)
// ---------------------------------------------------------------------------
__global__ __launch_bounds__(256) void zero_kernel(float* __restrict__ p, long n) {
    long i = (long)blockIdx.x * 256 + threadIdx.x;
    if (i < n) p[i] = 0.0f;
}

// ---------------------------------------------------------------------------
// Kernel 2: edge MLP layer 1.  Block = 16 edges, 256 threads = 8 waves,
// wave w owns output columns [16w, 16w+16).  Gathers h[col],h[row],edge_attr
// into LDS f16, runs K=288 WMMA loop, scaled-SiLU, stores x1 (f16) and
// accumulates BN sum / sumsq via LDS + global float atomics.
// ---------------------------------------------------------------------------
__global__ __launch_bounds__(256) void edge_mlp1_kernel(
    const float* __restrict__ h, const int* __restrict__ ei,
    const float* __restrict__ edge_attr,
    const _Float16* __restrict__ w1h, const float* __restrict__ eb1,
    _Float16* __restrict__ x1h, float* __restrict__ bn_sums, int E) {
    __shared__ __align__(32) _Float16 xin[16 * KP1];
    __shared__ float bns[DF];
    __shared__ float bnsq[DF];
    __shared__ int sc[16], sr[16];

    const int tid = threadIdx.x;
    const int eb  = blockIdx.x * 16;

    if (tid < 16) {
        sr[tid] = ei[eb + tid];          // edge_index[0] = row
        sc[tid] = ei[E + eb + tid];      // edge_index[1] = col
    }
    if (tid < DF) { bns[tid] = 0.0f; bnsq[tid] = 0.0f; }
    __syncthreads();

    // stage input tile [16][288] as f16 : concat(h[col], h[row], edge_attr, 0-pad)
    for (int idx = tid; idx < 16 * KP1; idx += 256) {
        int m = idx / KP1, k = idx - m * KP1;
        float v;
        if (k < DF)            v = h[(long)sc[m] * DF + k];
        else if (k < 2 * DF)   v = h[(long)sr[m] * DF + (k - DF)];
        else if (k < 2*DF+DE)  v = edge_attr[(long)(eb + m) * DE + (k - 2 * DF)];
        else                   v = 0.0f;
        xin[idx] = (_Float16)v;
    }
    __syncthreads();

    const int wave = tid >> 5, lane = tid & 31;
    const int nbase = wave * 16;
    v8f c = {};
#pragma unroll
    for (int k0 = 0; k0 < KP1; k0 += 32) {
        v16h a = load_a_frag(xin, lane & 15, KP1, k0, lane);
        v16h b = load_b_frag(w1h, nbase + (lane & 15), KP1, k0, lane);
        c = __builtin_amdgcn_wmma_f32_16x16x32_f16(false, a, false, b,
                                                   (short)0, c, false, false);
    }

    const int n    = nbase + (lane & 15);
    const float bi = eb1[n];
    const int mo   = (lane >> 4) * 8;
    float s = 0.0f, sq = 0.0f;
#pragma unroll
    for (int r = 0; r < 8; ++r) {
        float x = scaled_silu(c[r] + bi);
        x1h[(long)(eb + mo + r) * DF + n] = (_Float16)x;
        s += x; sq += x * x;
    }
    atomicAdd(&bns[n], s);
    atomicAdd(&bnsq[n], sq);
    __syncthreads();
    if (tid < DF) {
        atomicAdd(&bn_sums[tid], bns[tid]);
        atomicAdd(&bn_sums[DF + tid], bnsq[tid]);
    }
}

// ---------------------------------------------------------------------------
// Kernel 3: BN finalize -> per-feature scale/shift
// ---------------------------------------------------------------------------
__global__ __launch_bounds__(128) void bn_finalize_kernel(
    const float* __restrict__ bn_sums, const float* __restrict__ gamma,
    const float* __restrict__ beta, float* __restrict__ coef, float invE) {
    int k = threadIdx.x;
    float mean = bn_sums[k] * invE;
    float var  = bn_sums[DF + k] * invE - mean * mean;
    float sc   = gamma[k] * rsqrtf(var + BN_EPS);
    coef[k]      = sc;
    coef[DF + k] = beta[k] - mean * sc;
}

// ---------------------------------------------------------------------------
// Kernel 4: edge MLP layer 2 + attention gate + atomic scatter to agg.
// ---------------------------------------------------------------------------
__global__ __launch_bounds__(256) void edge_mlp2_kernel(
    const _Float16* __restrict__ x1h, const float* __restrict__ coef,
    const _Float16* __restrict__ w2h, const float* __restrict__ eb2,
    const float* __restrict__ aw, const float* __restrict__ ab,
    const int* __restrict__ ei, float* __restrict__ agg, int E) {
    __shared__ __align__(32) _Float16 xn[16 * DF];
    __shared__ float x2[16 * DF];
    __shared__ float att[16];
    __shared__ int sc[16];

    const int tid = threadIdx.x;
    const int eb  = blockIdx.x * 16;
    if (tid < 16) sc[tid] = ei[E + eb + tid];   // scatter target = col

    // load x1 tile, apply batch-norm, back to f16
    for (int idx = tid; idx < 16 * DF; idx += 256) {
        int k = idx & (DF - 1);
        float v = (float)x1h[(long)eb * DF + idx];
        xn[idx] = (_Float16)(v * coef[k] + coef[DF + k]);
    }
    __syncthreads();

    const int wave = tid >> 5, lane = tid & 31;
    const int nbase = wave * 16;
    v8f c = {};
#pragma unroll
    for (int k0 = 0; k0 < DF; k0 += 32) {
        v16h a = load_a_frag(xn, lane & 15, DF, k0, lane);
        v16h b = load_b_frag(w2h, nbase + (lane & 15), DF, k0, lane);
        c = __builtin_amdgcn_wmma_f32_16x16x32_f16(false, a, false, b,
                                                   (short)0, c, false, false);
    }
    const int n    = nbase + (lane & 15);
    const float bi = eb2[n];
    const int mo   = (lane >> 4) * 8;
#pragma unroll
    for (int r = 0; r < 8; ++r)
        x2[(mo + r) * DF + n] = scaled_silu(c[r] + bi);
    __syncthreads();

    if (tid < 16) {
        float s = ab[0];
#pragma unroll 4
        for (int k = 0; k < DF; ++k) s += x2[tid * DF + k] * aw[k];
        att[tid] = 1.0f / (1.0f + __expf(-s));
    }
    __syncthreads();

    for (int idx = tid; idx < 16 * DF; idx += 256) {
        int m = idx >> 7, k = idx & (DF - 1);
        atomicAdd(&agg[(long)sc[m] * DF + k], x2[idx] * att[m]);
    }
}

// ---------------------------------------------------------------------------
// Kernel 5: node MLP with residual.  Block = 16 nodes, 8 waves.
// z1 = scaled_silu(nw1 @ [h, agg/100]); out = h + nw2 @ z1 + nb2
// ---------------------------------------------------------------------------
__global__ __launch_bounds__(256) void node_mlp_kernel(
    const float* __restrict__ h, const float* __restrict__ agg,
    const _Float16* __restrict__ nw1h, const float* __restrict__ nb1,
    const _Float16* __restrict__ nw2h, const float* __restrict__ nb2,
    float* __restrict__ out) {
    __shared__ __align__(32) _Float16 zin[16 * 2 * DF];
    __shared__ __align__(32) _Float16 z1[16 * DF];

    const int tid = threadIdx.x;
    const int nb0 = blockIdx.x * 16;

    for (int idx = tid; idx < 16 * 2 * DF; idx += 256) {
        int m = idx >> 8, k = idx & (2 * DF - 1);
        float v = (k < DF) ? h[(long)(nb0 + m) * DF + k]
                           : agg[(long)(nb0 + m) * DF + (k - DF)] * INV_NORM;
        zin[idx] = (_Float16)v;
    }
    __syncthreads();

    const int wave = tid >> 5, lane = tid & 31;
    const int nbase = wave * 16;
    const int n  = nbase + (lane & 15);
    const int mo = (lane >> 4) * 8;

    v8f c = {};
#pragma unroll
    for (int k0 = 0; k0 < 2 * DF; k0 += 32) {
        v16h a = load_a_frag(zin, lane & 15, 2 * DF, k0, lane);
        v16h b = load_b_frag(nw1h, n, 2 * DF, k0, lane);
        c = __builtin_amdgcn_wmma_f32_16x16x32_f16(false, a, false, b,
                                                   (short)0, c, false, false);
    }
    const float b1 = nb1[n];
#pragma unroll
    for (int r = 0; r < 8; ++r)
        z1[(mo + r) * DF + n] = (_Float16)scaled_silu(c[r] + b1);
    __syncthreads();

    v8f c2 = {};
#pragma unroll
    for (int k0 = 0; k0 < DF; k0 += 32) {
        v16h a = load_a_frag(z1, lane & 15, DF, k0, lane);
        v16h b = load_b_frag(nw2h, n, DF, k0, lane);
        c2 = __builtin_amdgcn_wmma_f32_16x16x32_f16(false, a, false, b,
                                                    (short)0, c2, false, false);
    }
    const float b2 = nb2[n];
#pragma unroll
    for (int r = 0; r < 8; ++r) {
        long o = (long)(nb0 + mo + r) * DF + n;
        out[o] = h[o] + c2[r] + b2;
    }
}

// ---------------------------------------------------------------------------
// host launcher
// ---------------------------------------------------------------------------
extern "C" void kernel_launch(void* const* d_in, const int* in_sizes, int n_in,
                              void* d_out, int out_size, void* d_ws, size_t ws_size,
                              hipStream_t stream) {
    const float* h         = (const float*)d_in[0];
    const int*   ei        = (const int*)  d_in[1];
    const float* edge_attr = (const float*)d_in[2];
    const float* ew1       = (const float*)d_in[3];
    const float* eb1       = (const float*)d_in[4];
    const float* gamma     = (const float*)d_in[5];
    const float* beta      = (const float*)d_in[6];
    const float* ew2       = (const float*)d_in[7];
    const float* eb2       = (const float*)d_in[8];
    const float* aw        = (const float*)d_in[9];
    const float* ab        = (const float*)d_in[10];
    const float* nw1       = (const float*)d_in[11];
    const float* nb1       = (const float*)d_in[12];
    const float* nw2       = (const float*)d_in[13];
    const float* nb2       = (const float*)d_in[14];
    float* out = (float*)d_out;

    const int N = in_sizes[0] / DF;   // 50000 nodes
    const int E = in_sizes[2] / DE;   // 800000 edges

    // workspace layout (all offsets 256B-aligned)
    char* ws = (char*)d_ws;
    _Float16* ew1h = (_Float16*)(ws + 0);            // 128*288*2 = 73728
    _Float16* ew2h = (_Float16*)(ws + 73728);        // 128*128*2 = 32768
    _Float16* nw1h = (_Float16*)(ws + 106496);       // 128*256*2 = 65536
    _Float16* nw2h = (_Float16*)(ws + 172032);       // 128*128*2 = 32768
    float*    coef = (float*)   (ws + 204800);       // 256 floats (scale,shift)
    float*    sums = (float*)   (ws + 205824);       // 256 floats (sum, sumsq)
    float*    agg  = (float*)   (ws + 206848);       // N*128 floats (contig after sums)
    _Float16* x1h  = (_Float16*)(ws + 206848 + (size_t)N * DF * 4);  // E*128 halves

    cvt_weights_kernel<<<(DF * KP1 + 255) / 256, 256, 0, stream>>>(
        ew1, ew2, nw1, nw2, ew1h, ew2h, nw1h, nw2h);

    long nzero = 2 * DF + (long)N * DF;   // sums + agg (contiguous)
    zero_kernel<<<(int)((nzero + 255) / 256), 256, 0, stream>>>(sums, nzero);

    edge_mlp1_kernel<<<E / 16, 256, 0, stream>>>(
        h, ei, edge_attr, ew1h, eb1, x1h, sums, E);

    bn_finalize_kernel<<<1, DF, 0, stream>>>(sums, gamma, beta, coef, 1.0f / (float)E);

    edge_mlp2_kernel<<<E / 16, 256, 0, stream>>>(
        x1h, coef, ew2h, eb2, aw, ab, ei, agg, E);

    node_mlp_kernel<<<N / 16, 256, 0, stream>>>(
        h, agg, nw1h, nb1, nw2h, nb2, out);
}
